// EdgeMessagePassing_78142634983851
// MI455X (gfx1250) — matrix operand order repaired and checked by reading, not compiled, not run on hardware
//
#include <hip/hip_runtime.h>
#include <math.h>

typedef __attribute__((ext_vector_type(16))) _Float16 v16h;
typedef __attribute__((ext_vector_type(8)))  float    v8f;

#define DEVINL __device__ __forceinline__

constexpr int N_NODES = 50000;
constexpr int N_EDGES = 800000;
constexpr int ND  = 64;    // node feature dim
constexpr int ED  = 64;    // edge feature dim
constexpr int HID = 128;   // hidden dim
constexpr float LNEPS = 1e-5f;

// ---------------------------------------------------------------------------
// WMMA fragment layouts (wave32, 16x16x32 f16, per CDNA5 ISA):
//  A (16x32, f16): lane m=lane&15 is row M; element i holds
//      k = (i&7) + (i>=8 ? 16 : 0) + (lane>=16 ? 8 : 0)
//    -> per lane: two contiguous 8-element runs at k0 and k0+16.
//  B (32x16, f16): column N = lane&15; element i holds
//      k = (i&7) + (i>=8 ? 16 : 0) + (lane>=16 ? 8 : 0)  (same as A)
//  C/D (16x16, f32): element r holds row M = r + (lane>=16 ? 8 : 0),
//      column N = lane&15.
// ---------------------------------------------------------------------------

DEVINL void atomic_add_f32(float* p, float v) { unsafeAtomicAdd(p, v); }

DEVINL float gelu_exact(float x) {
    return 0.5f * x * (1.0f + erff(x * 0.70710678118654752440f));
}

// Load an A fragment from a row pointer: two 8-float runs at p[0..7], p[16..23]
DEVINL v16h load_a_f32(const float* __restrict__ p) {
    v16h a;
#pragma unroll
    for (int j = 0; j < 8; ++j) a[j]     = (_Float16)p[j];
#pragma unroll
    for (int j = 0; j < 8; ++j) a[8 + j] = (_Float16)p[16 + j];
    return a;
}

DEVINL v16h load_a_f32_scaled(const float* __restrict__ p, float sc) {
    v16h a;
#pragma unroll
    for (int j = 0; j < 8; ++j) a[j]     = (_Float16)(p[j] * sc);
#pragma unroll
    for (int j = 0; j < 8; ++j) a[8 + j] = (_Float16)(p[16 + j] * sc);
    return a;
}

DEVINL v8f splat8(float v) {
    v8f r;
#pragma unroll
    for (int i = 0; i < 8; ++i) r[i] = v;
    return r;
}

// ---------------------------------------------------------------------------
// Prep: repack fp32 weight matrix W[K][N] (row-major) into f16 B-fragment-major
// layout: out[frag*512 + lane*16 + i] with frag = kt*(N/16)+nt,
//   value = W[kt*32 + (i&7) + (i>=8?16:0) + (lane>=16?8:0)][nt*16 + (lane&15)]
// Total elements == K*N, and tid == frag*512 + lane*16 + i.
// ---------------------------------------------------------------------------
__global__ void pack_weights_kernel(const float* __restrict__ W, int K, int N,
                                    _Float16* __restrict__ out) {
    int tid = blockIdx.x * blockDim.x + threadIdx.x;
    if (tid >= K * N) return;
    int i    = tid & 15;
    int lane = (tid >> 4) & 31;
    int frag = tid >> 9;
    int NT   = N >> 4;
    int kt   = frag / NT;
    int nt   = frag - kt * NT;
    int k    = kt * 32 + (i & 7) + ((i >= 8) ? 16 : 0) + ((lane >= 16) ? 8 : 0);
    int col  = nt * 16 + (lane & 15);
    out[tid] = (_Float16)W[k * N + col];
}

__global__ void zero_kernel(float* __restrict__ p, int n) {
    int i = blockIdx.x * blockDim.x + threadIdx.x;
    for (; i < n; i += gridDim.x * blockDim.x) p[i] = 0.0f;
}

// ---------------------------------------------------------------------------
// Edge kernel: 8 waves/block, 1 wave = 16 edges.
//   x = [nf[src] | nf[dst] | ef]  (192)  -> GELU(x@W1+b1) (128) -> @W2+b2 (64)
//   edge_out = LN(ef + update); scatter-add edge_out into agg[dst], cnt[dst].
// ---------------------------------------------------------------------------
__global__ __launch_bounds__(256) void edge_kernel(
    const float* __restrict__ nf, const int* __restrict__ ei,
    const float* __restrict__ ef,
    const _Float16* __restrict__ w1, const float* __restrict__ b1,
    const _Float16* __restrict__ w2, const float* __restrict__ b2,
    const float* __restrict__ gamma, const float* __restrict__ beta,
    float* __restrict__ eout, float* __restrict__ agg, float* __restrict__ cnt)
{
    __shared__ _Float16 stage_s[8][16 * HID];   // 4KB per wave (32KB/block)

    const int lane = threadIdx.x & 31;
    const int wv   = threadIdx.x >> 5;
    const int m    = lane & 15;          // row within tile
    const int hi   = lane >> 4;          // K/row-half selector
    const int e0   = (blockIdx.x * 8 + wv) * 16;
    const int edge = e0 + m;

    const int src = ei[edge];
    const int dst = ei[N_EDGES + edge];

    const float* base_src = nf + (size_t)src  * ND;
    const float* base_dst = nf + (size_t)dst  * ND;
    const float* base_ef  = ef + (size_t)edge * ED;

    // ---- GEMM1: [16x192] @ [192x128], K tiles kt=0..5, N tiles nt=0..7 ----
    v8f acc[8];
#pragma unroll
    for (int nt = 0; nt < 8; ++nt) acc[nt] = splat8(b1[nt * 16 + m]);

#pragma unroll
    for (int kt = 0; kt < 6; ++kt) {
        const float* bp = (kt < 2) ? base_src : (kt < 4) ? base_dst : base_ef;
        v16h a = load_a_f32(bp + (kt & 1) * 32 + hi * 8);
#pragma unroll
        for (int nt = 0; nt < 8; ++nt) {
            v16h b = *(const v16h*)(w1 + (size_t)(kt * 8 + nt) * 512 + lane * 16);
            acc[nt] = __builtin_amdgcn_wmma_f32_16x16x32_f16(
                false, a, false, b, (short)0, acc[nt], false, false);
        }
    }

    // ---- GELU + transpose C-layout -> row-major f16 staging in LDS ----
    _Float16* st = stage_s[wv];
#pragma unroll
    for (int nt = 0; nt < 8; ++nt) {
        int c = nt * 16 + m;
#pragma unroll
        for (int r = 0; r < 8; ++r) {
            st[(r + hi * 8) * HID + c] = (_Float16)gelu_exact(acc[nt][r]);
        }
    }

    // ---- GEMM2: [16x128] @ [128x64], kt2=0..3, nt2=0..3 ----
    v8f acc2[4];
#pragma unroll
    for (int nt = 0; nt < 4; ++nt) acc2[nt] = splat8(b2[nt * 16 + m]);

#pragma unroll
    for (int kt = 0; kt < 4; ++kt) {
        const _Float16* pr = st + m * HID + kt * 32 + hi * 8;
        v16h a;
#pragma unroll
        for (int j = 0; j < 8; ++j) { a[j] = pr[j]; a[8 + j] = pr[16 + j]; }
#pragma unroll
        for (int nt = 0; nt < 4; ++nt) {
            v16h b = *(const v16h*)(w2 + (size_t)(kt * 4 + nt) * 512 + lane * 16);
            acc2[nt] = __builtin_amdgcn_wmma_f32_16x16x32_f16(
                false, a, false, b, (short)0, acc2[nt], false, false);
        }
    }

    // ---- stash MLP output (pre-residual) as f32 in the same LDS buffer ----
    float* stf = (float*)st;   // 16 x 64 f32 == same 4KB footprint
#pragma unroll
    for (int nt = 0; nt < 4; ++nt) {
        int c = nt * 16 + m;
#pragma unroll
        for (int r = 0; r < 8; ++r) stf[(r + hi * 8) * ED + c] = acc2[nt][r];
    }

    // ---- residual + LayerNorm; lane = (row m, half hi: 32 cols) ----
    float v[32];
    const float* res = base_ef + hi * 32;
    float s = 0.0f, q = 0.0f;
#pragma unroll
    for (int j = 0; j < 32; ++j) {
        float x = stf[m * ED + hi * 32 + j] + res[j];
        v[j] = x; s += x; q += x * x;
    }
    s += __shfl_xor(s, 16, 32);
    q += __shfl_xor(q, 16, 32);
    float mean = s * (1.0f / 64.0f);
    float var  = q * (1.0f / 64.0f) - mean * mean;
    float rstd = rsqrtf(var + LNEPS);

    float* op = eout + (size_t)edge * ED + hi * 32;
    float* ap = agg  + (size_t)dst  * ND + hi * 32;
#pragma unroll
    for (int j = 0; j < 32; ++j) {
        float o = (v[j] - mean) * rstd * gamma[hi * 32 + j] + beta[hi * 32 + j];
        op[j] = o;
        atomic_add_f32(ap + j, o);
    }
    if (!hi) atomic_add_f32(cnt + dst, 1.0f);
}

// ---------------------------------------------------------------------------
// Node kernel: 1 wave = 16 nodes.
//   x = [nf | agg/max(cnt,1)] (128) -> GELU(x@W1+b1) (128) -> @W2+b2 (64)
//   node_out = LN(nf + update)
// ---------------------------------------------------------------------------
__global__ __launch_bounds__(256) void node_kernel(
    const float* __restrict__ nf,
    const _Float16* __restrict__ w1, const float* __restrict__ b1,
    const _Float16* __restrict__ w2, const float* __restrict__ b2,
    const float* __restrict__ gamma, const float* __restrict__ beta,
    const float* __restrict__ agg, const float* __restrict__ cnt,
    float* __restrict__ nout)
{
    __shared__ _Float16 stage_s[8][16 * HID];

    const int lane = threadIdx.x & 31;
    const int wv   = threadIdx.x >> 5;
    const int m    = lane & 15;
    const int hi   = lane >> 4;
    const int t0   = (blockIdx.x * 8 + wv) * 16;
    if (t0 >= N_NODES) return;              // wave-uniform
    const int node = t0 + m;

    const float inv = 1.0f / fmaxf(cnt[node], 1.0f);
    const float* base_nf  = nf  + (size_t)node * ND;
    const float* base_agg = agg + (size_t)node * ND;

    v8f acc[8];
#pragma unroll
    for (int nt = 0; nt < 8; ++nt) acc[nt] = splat8(b1[nt * 16 + m]);

#pragma unroll
    for (int kt = 0; kt < 4; ++kt) {
        v16h a;
        if (kt < 2) a = load_a_f32(base_nf + (kt & 1) * 32 + hi * 8);
        else        a = load_a_f32_scaled(base_agg + (kt & 1) * 32 + hi * 8, inv);
#pragma unroll
        for (int nt = 0; nt < 8; ++nt) {
            v16h b = *(const v16h*)(w1 + (size_t)(kt * 8 + nt) * 512 + lane * 16);
            acc[nt] = __builtin_amdgcn_wmma_f32_16x16x32_f16(
                false, a, false, b, (short)0, acc[nt], false, false);
        }
    }

    _Float16* st = stage_s[wv];
#pragma unroll
    for (int nt = 0; nt < 8; ++nt) {
        int c = nt * 16 + m;
#pragma unroll
        for (int r = 0; r < 8; ++r)
            st[(r + hi * 8) * HID + c] = (_Float16)gelu_exact(acc[nt][r]);
    }

    v8f acc2[4];
#pragma unroll
    for (int nt = 0; nt < 4; ++nt) acc2[nt] = splat8(b2[nt * 16 + m]);

#pragma unroll
    for (int kt = 0; kt < 4; ++kt) {
        const _Float16* pr = st + m * HID + kt * 32 + hi * 8;
        v16h a;
#pragma unroll
        for (int j = 0; j < 8; ++j) { a[j] = pr[j]; a[8 + j] = pr[16 + j]; }
#pragma unroll
        for (int nt = 0; nt < 4; ++nt) {
            v16h b = *(const v16h*)(w2 + (size_t)(kt * 4 + nt) * 512 + lane * 16);
            acc2[nt] = __builtin_amdgcn_wmma_f32_16x16x32_f16(
                false, a, false, b, (short)0, acc2[nt], false, false);
        }
    }

    float* stf = (float*)st;
#pragma unroll
    for (int nt = 0; nt < 4; ++nt) {
        int c = nt * 16 + m;
#pragma unroll
        for (int r = 0; r < 8; ++r) stf[(r + hi * 8) * ND + c] = acc2[nt][r];
    }

    float v[32];
    const float* res = base_nf + hi * 32;
    float s = 0.0f, q = 0.0f;
#pragma unroll
    for (int j = 0; j < 32; ++j) {
        float x = stf[m * ND + hi * 32 + j] + res[j];
        v[j] = x; s += x; q += x * x;
    }
    s += __shfl_xor(s, 16, 32);
    q += __shfl_xor(q, 16, 32);
    float mean = s * (1.0f / 64.0f);
    float var  = q * (1.0f / 64.0f) - mean * mean;
    float rstd = rsqrtf(var + LNEPS);

    float* op = nout + (size_t)node * ND + hi * 32;
#pragma unroll
    for (int j = 0; j < 32; ++j) {
        op[j] = (v[j] - mean) * rstd * gamma[hi * 32 + j] + beta[hi * 32 + j];
    }
}

// ---------------------------------------------------------------------------
extern "C" void kernel_launch(void* const* d_in, const int* in_sizes, int n_in,
                              void* d_out, int out_size, void* d_ws, size_t ws_size,
                              hipStream_t stream) {
    const float* nf   = (const float*)d_in[0];
    const int*   ei   = (const int*)  d_in[1];
    const float* ef   = (const float*)d_in[2];
    const float* e_w1 = (const float*)d_in[3];
    const float* e_b1 = (const float*)d_in[4];
    const float* e_w2 = (const float*)d_in[5];
    const float* e_b2 = (const float*)d_in[6];
    const float* n_w1 = (const float*)d_in[7];
    const float* n_b1 = (const float*)d_in[8];
    const float* n_w2 = (const float*)d_in[9];
    const float* n_b2 = (const float*)d_in[10];
    const float* e_g  = (const float*)d_in[11];
    const float* e_b  = (const float*)d_in[12];
    const float* n_g  = (const float*)d_in[13];
    const float* n_b  = (const float*)d_in[14];

    float* nout = (float*)d_out;
    float* eout = nout + (size_t)N_NODES * ND;

    // Workspace layout: agg (N*64 f32) | cnt (N f32) | f16 weight fragments
    float* agg = (float*)d_ws;
    float* cnt = agg + (size_t)N_NODES * ND;
    _Float16* ew1f = (_Float16*)(cnt + N_NODES);    // 192*128 (32B-aligned)
    _Float16* ew2f = ew1f + 192 * 128;              // 128*64
    _Float16* nw1f = ew2f + 128 * 64;               // 128*128
    _Float16* nw2f = nw1f + 128 * 128;              // 128*64

    const int zn = N_NODES * ND + N_NODES;
    zero_kernel<<<(zn + 255) / 256, 256, 0, stream>>>((float*)d_ws, zn);

    pack_weights_kernel<<<(192 * 128 + 255) / 256, 256, 0, stream>>>(e_w1, 192, 128, ew1f);
    pack_weights_kernel<<<(128 * 64  + 255) / 256, 256, 0, stream>>>(e_w2, 128, 64,  ew2f);
    pack_weights_kernel<<<(128 * 128 + 255) / 256, 256, 0, stream>>>(n_w1, 128, 128, nw1f);
    pack_weights_kernel<<<(128 * 64  + 255) / 256, 256, 0, stream>>>(n_w2, 128, 64,  nw2f);

    // 800000 edges / (8 waves * 16 edges) = 6250 blocks (exact)
    edge_kernel<<<6250, 256, 0, stream>>>(nf, ei, ef, ew1f, e_b1, ew2f, e_b2,
                                          e_g, e_b, eout, agg, cnt);

    // 50000 nodes / 16 = 3125 tiles; 8 tiles/block -> 391 blocks
    node_kernel<<<391, 256, 0, stream>>>(nf, nw1f, n_b1, nw2f, n_b2,
                                         n_g, n_b, agg, cnt, nout);
}
